// CKGConvBlock_61160334295117
// MI455X (gfx1250) — matrix operand order repaired and checked by reading, not compiled, not run on hardware
//
#include <hip/hip_runtime.h>
#include <math.h>

#define N_NODES 100000
#define N_EDGES 800000
#define D_NODE 128
#define D_PE   16
#define D_NF   144
#define D_EDGE 16
#define D_EF   32
#define D_MOD  64
#define D_OUT  128
#define D_FFN  512
#define EPSBN  1e-5f

typedef __attribute__((ext_vector_type(16))) _Float16 v16h;
typedef __attribute__((ext_vector_type(8)))  _Float16 v8h;
typedef __attribute__((ext_vector_type(8)))  float    v8f;
typedef _Float16 half_t;

#define WMMA_F16(a, b, c) __builtin_amdgcn_wmma_f32_16x16x32_f16(false, (a), false, (b), (short)0, (c), false, false)

__device__ __forceinline__ float gelu_exact(float x) {
  return 0.5f * x * (1.0f + erff(x * 0.70710678118654752f));
}

// A fragment: 16x32 f16 tile, row-major source with stride lda (halves).
// Lane L holds row M=L&15; halves j=0..7 -> K=j+kb, j=8..15 -> K=j+8+kb, kb=(L>=16)?8:0.
// Both 8-half runs are contiguous and 16B-aligned -> ds_load_b128 x2.
__device__ __forceinline__ v16h load_A_lds(const half_t* A, int lda, int lane) {
  int m  = lane & 15;
  int kb = (lane >> 4) << 3;
  const half_t* row = A + m * lda + kb;
  v8h lo = *(const v8h*)(row);        // K = kb .. kb+7
  v8h hi = *(const v8h*)(row + 16);   // K = kb+16 .. kb+23
  v16h a;
#pragma unroll
  for (int j = 0; j < 8; ++j) { a[j] = lo[j]; a[j + 8] = hi[j]; }
  return a;
}

// B fragment from pre-packed fragment buffer: tile = kt*ntiles_n + nt,
// per tile 32 lanes x 16 contiguous halves (1KB). One 32B load per lane.
__device__ __forceinline__ v16h load_frag(const half_t* F, int tile, int lane) {
  return *((const v16h*)(F + (size_t)tile * 512) + lane);
}

// Pack row-major KxN f32 weights into WMMA-B fragment layout (f16), zero-filling K/N overhang.
// Fragment element j of lane L = W[kt*32 + ((L>=16)?8:0) + (j&7) + ((j>=8)?16:0)][nt*16 + (L&15)].
__global__ void k_pack_frags(const float* __restrict__ W, half_t* __restrict__ F,
                             int K, int N, int ntiles_n, int ntiles_total) {
  int idx  = blockIdx.x * blockDim.x + threadIdx.x;
  int tile = idx >> 5;
  if (tile >= ntiles_total) return;
  int lane = idx & 31;
  int kt = tile / ntiles_n, nt = tile % ntiles_n;
  int n  = nt * 16 + (lane & 15);
  int kb = kt * 32 + ((lane >> 4) << 3);
  half_t* dst = F + (size_t)tile * 512 + lane * 16;
#pragma unroll
  for (int j = 0; j < 16; ++j) {
    int k = kb + ((j >> 3) << 4) + (j & 7);
    float v = (k < K && n < N) ? W[(size_t)k * N + n] : 0.0f;
    dst[j] = (half_t)v;
  }
}

// ---------------- Edge pipeline: modulator MLP (WMMA) + gather/modulate/scatter ---------------
__global__ __launch_bounds__(128)
void k_edge(const float* __restrict__ x, const float* __restrict__ x_pe,
            const float* __restrict__ ea, const float* __restrict__ ep,
            const long long* __restrict__ eidx,
            const half_t* __restrict__ Wm1f, const float* __restrict__ bm1,
            const half_t* __restrict__ Wm2f, const float* __restrict__ bm2,
            float* __restrict__ summed, float* __restrict__ cnt, float* __restrict__ deg) {
  __shared__ half_t s_hid[4][16][D_MOD];
  __shared__ float  s_mod[4][16][D_NF + 4];
  int lane = threadIdx.x & 31;
  int w    = threadIdx.x >> 5;
  int e0   = (blockIdx.x * 4 + w) * 16;

  __builtin_prefetch(Wm2f, 0, 1);   // global_prefetch_b8: warm WGP$/L2 with layer-2 fragments

  // A fragment: edge-feature tile ec = concat(edge_attr, edge_pe), [16 x 32], f32 -> f16
  v16h a;
  {
    int m  = lane & 15;
    int kb = (lane >> 4) << 3;
    int e  = e0 + m; if (e >= N_EDGES) e = N_EDGES - 1;
#pragma unroll
    for (int j = 0; j < 16; ++j) {
      int k = ((j >> 3) << 4) + (j & 7) + kb;
      float v = (k < D_EDGE) ? ea[(long long)e * D_EDGE + k]
                             : ep[(long long)e * D_PE + (k - D_EDGE)];
      a[j] = (half_t)v;
    }
  }
  // Layer 1: hidden = gelu(ec @ Wm1 + b1), K=32 exact, N=64 -> 4 tiles
#pragma unroll
  for (int t = 0; t < 4; ++t) {
    v8f c = {};
    c = WMMA_F16(a, load_frag(Wm1f, t, lane), c);
    int   n    = t * 16 + (lane & 15);
    float bias = bm1[n];
#pragma unroll
    for (int v = 0; v < 8; ++v) {
      int m = v + ((lane >> 4) << 3);
      s_hid[w][m][n] = (half_t)gelu_exact(c[v] + bias);
    }
  }
  __syncthreads();
  // Layer 2: mod = hidden @ Wm2 + b2, K=64 -> 2 chunks, N=144 -> 9 tiles
  v16h a0 = load_A_lds(&s_hid[w][0][0],  D_MOD, lane);
  v16h a1 = load_A_lds(&s_hid[w][0][32], D_MOD, lane);
#pragma unroll
  for (int t = 0; t < 9; ++t) {
    v8f c = {};
    c = WMMA_F16(a0, load_frag(Wm2f, t,     lane), c);
    c = WMMA_F16(a1, load_frag(Wm2f, 9 + t, lane), c);
    int   n    = t * 16 + (lane & 15);
    float bias = bm2[n];
#pragma unroll
    for (int v = 0; v < 8; ++v) {
      int m = v + ((lane >> 4) << 3);
      s_mod[w][m][n] = c[v] + bias;
    }
  }
  __syncthreads();
  // Scatter: msg = xc[src] * mod -> atomic mean-accumulate at dst; degree counts
  const long long* srcp = eidx;
  const long long* dstp = eidx + N_EDGES;
  for (int m = 0; m < 16; ++m) {
    int e = e0 + m;
    if (e >= N_EDGES) break;
    long long s = srcp[e], d = dstp[e];
    for (int f = lane; f < D_NF; f += 32) {
      float xv = (f < D_NODE) ? x[s * D_NODE + f] : x_pe[s * D_PE + (f - D_NODE)];
      atomicAdd(&summed[d * D_NF + f], xv * s_mod[w][m][f]);
    }
    if (lane == 0) { atomicAdd(&cnt[d], 1.0f); atomicAdd(&deg[s], 1.0f); }
  }
}

// ---------------- Node linear: agg @ W_lin + degree scaling + BN1 stats ----------------
__global__ __launch_bounds__(128)
void k_node_lin(const float* __restrict__ summed, const float* __restrict__ cnt,
                const float* __restrict__ deg,
                const half_t* __restrict__ Wlnf, const float* __restrict__ blin,
                const float* __restrict__ th1, const float* __restrict__ th2,
                float* __restrict__ outbuf, float* __restrict__ sum1, float* __restrict__ sq1) {
  __shared__ half_t s_agg[4][16][160];   // K padded 144 -> 160 with zeros
  int lane = threadIdx.x & 31;
  int w    = threadIdx.x >> 5;
  int n0   = (blockIdx.x * 4 + w) * 16;
  // Stage agg = summed/max(cnt,1) as f16, float4 loads (36 vec4 per row + 4 zero pads)
  for (int i = lane; i < 16 * 40; i += 32) {
    int r = i / 40, q = i % 40;
    int node = n0 + r;
    float4 v = make_float4(0.f, 0.f, 0.f, 0.f);
    if (node < N_NODES && q < 36) {
      v = *(const float4*)&summed[(long long)node * D_NF + q * 4];
      float inv = 1.0f / fmaxf(cnt[node], 1.0f);
      v.x *= inv; v.y *= inv; v.z *= inv; v.w *= inv;
    }
    half_t* dst = &s_agg[w][r][q * 4];
    dst[0] = (half_t)v.x; dst[1] = (half_t)v.y; dst[2] = (half_t)v.z; dst[3] = (half_t)v.w;
  }
  __syncthreads();
  v16h af[5];
#pragma unroll
  for (int kc = 0; kc < 5; ++kc) af[kc] = load_A_lds(&s_agg[w][0][kc * 32], 160, lane);
#pragma unroll
  for (int t = 0; t < 8; ++t) {
    v8f c = {};
#pragma unroll
    for (int kc = 0; kc < 5; ++kc)
      c = WMMA_F16(af[kc], load_frag(Wlnf, kc * 8 + t, lane), c);
    int n = t * 16 + (lane & 15);
#pragma unroll
    for (int v = 0; v < 8; ++v) {
      int node = n0 + v + ((lane >> 4) << 3);
      if (node < N_NODES) {
        float o  = c[v] + blin[n];
        float ds = sqrtf(fmaxf(deg[node], 1.0f));
        o = o * th1[n] + ds * (o * th2[n]);
        outbuf[(long long)node * D_OUT + n] = o;
        atomicAdd(&sum1[n], o);
        atomicAdd(&sq1[n], o * o);
      }
    }
  }
}

__global__ void k_stats(const float* __restrict__ sum, const float* __restrict__ sq,
                        float* __restrict__ mu, float* __restrict__ rs, float count) {
  int i = threadIdx.x;
  if (i < D_OUT) {
    float m = sum[i] / count;
    mu[i] = m;
    rs[i] = rsqrtf(sq[i] / count - m * m + EPSBN);
  }
}

__global__ void k_bn1(float* __restrict__ hbuf, const float* __restrict__ x,
                      const float* __restrict__ mu, const float* __restrict__ rs,
                      const float* __restrict__ g, const float* __restrict__ b) {
  long long i = (long long)blockIdx.x * blockDim.x + threadIdx.x;
  if (i < (long long)N_NODES * D_OUT) {
    int f = (int)(i & (D_OUT - 1));
    hbuf[i] = (hbuf[i] - mu[f]) * rs[f] * g[f] + b[f] + x[i];
  }
}

// ---------------- FFN: gelu(h @ Wf1 + b1) @ Wf2 + b2, fused K-streaming ----------------
__global__ __launch_bounds__(128)
void k_ffn(const float* __restrict__ hbuf,
           const half_t* __restrict__ Wf1f, const float* __restrict__ bf1,
           const half_t* __restrict__ Wf2f, const float* __restrict__ bf2,
           float* __restrict__ ybuf, float* __restrict__ sum2, float* __restrict__ sq2) {
  __shared__ half_t s_h[4][16][D_OUT];
  __shared__ half_t s_g[4][16][32];
  int lane = threadIdx.x & 31;
  int w    = threadIdx.x >> 5;
  int n0   = (blockIdx.x * 4 + w) * 16;
  // Stage h tile as f16, float4 loads (32 vec4 per row)
  for (int i = lane; i < 16 * 32; i += 32) {
    int r = i >> 5, q = i & 31;
    int node = n0 + r;
    float4 v = make_float4(0.f, 0.f, 0.f, 0.f);
    if (node < N_NODES) v = *(const float4*)&hbuf[(long long)node * D_OUT + q * 4];
    half_t* dst = &s_h[w][r][q * 4];
    dst[0] = (half_t)v.x; dst[1] = (half_t)v.y; dst[2] = (half_t)v.z; dst[3] = (half_t)v.w;
  }
  __syncthreads();
  v16h ah[4];
#pragma unroll
  for (int kc = 0; kc < 4; ++kc) ah[kc] = load_A_lds(&s_h[w][0][kc * 32], D_OUT, lane);
  v8f acc[8];
#pragma unroll
  for (int t = 0; t < 8; ++t) { v8f z = {}; acc[t] = z; }
  for (int hc = 0; hc < 16; ++hc) {          // 512 hidden cols in chunks of 32
#pragma unroll
    for (int tt = 0; tt < 2; ++tt) {
      int nt = hc * 2 + tt;                  // hidden 16-col tile index
      v8f c = {};
#pragma unroll
      for (int kc = 0; kc < 4; ++kc)
        c = WMMA_F16(ah[kc], load_frag(Wf1f, kc * 32 + nt, lane), c);
      int   n    = lane & 15;
      float bias = bf1[nt * 16 + n];
#pragma unroll
      for (int v = 0; v < 8; ++v) {
        int m = v + ((lane >> 4) << 3);
        s_g[w][m][tt * 16 + n] = (half_t)gelu_exact(c[v] + bias);
      }
    }
    __syncthreads();
    v16h a2 = load_A_lds(&s_g[w][0][0], 32, lane);
#pragma unroll
    for (int t = 0; t < 8; ++t)
      acc[t] = WMMA_F16(a2, load_frag(Wf2f, hc * 8 + t, lane), acc[t]);
    __syncthreads();
  }
#pragma unroll
  for (int t = 0; t < 8; ++t) {
    int   n    = t * 16 + (lane & 15);
    float bias = bf2[n];
#pragma unroll
    for (int v = 0; v < 8; ++v) {
      int node = n0 + v + ((lane >> 4) << 3);
      if (node < N_NODES) {
        float y = hbuf[(long long)node * D_OUT + n] + acc[t][v] + bias;
        ybuf[(long long)node * D_OUT + n] = y;
        atomicAdd(&sum2[n], y);
        atomicAdd(&sq2[n], y * y);
      }
    }
  }
}

__global__ void k_bn2(const float* __restrict__ y, float* __restrict__ out,
                      const float* __restrict__ mu, const float* __restrict__ rs,
                      const float* __restrict__ g, const float* __restrict__ b) {
  long long i = (long long)blockIdx.x * blockDim.x + threadIdx.x;
  if (i < (long long)N_NODES * D_OUT) {
    int f = (int)(i & (D_OUT - 1));
    out[i] = (y[i] - mu[f]) * rs[f] * g[f] + b[f];
  }
}

extern "C" void kernel_launch(void* const* d_in, const int* in_sizes, int n_in,
                              void* d_out, int out_size, void* d_ws, size_t ws_size,
                              hipStream_t stream) {
  (void)in_sizes; (void)n_in; (void)out_size; (void)ws_size;
  const float*     x    = (const float*)d_in[0];
  const float*     x_pe = (const float*)d_in[1];
  const float*     ea   = (const float*)d_in[2];
  const float*     ep   = (const float*)d_in[3];
  const long long* eidx = (const long long*)d_in[4];
  const float* Wm1 = (const float*)d_in[5];
  const float* bm1 = (const float*)d_in[6];
  const float* Wm2 = (const float*)d_in[7];
  const float* bm2 = (const float*)d_in[8];
  const float* Wln = (const float*)d_in[9];
  const float* bln = (const float*)d_in[10];
  const float* th1 = (const float*)d_in[11];
  const float* th2 = (const float*)d_in[12];
  const float* g1  = (const float*)d_in[13];
  const float* be1 = (const float*)d_in[14];
  const float* g2  = (const float*)d_in[15];
  const float* be2 = (const float*)d_in[16];
  const float* Wf1 = (const float*)d_in[17];
  const float* bf1 = (const float*)d_in[18];
  const float* Wf2 = (const float*)d_in[19];
  const float* bf2 = (const float*)d_in[20];

  // Fragment-tile counts (K-tiles x N-tiles)
  const int T_M1 = 1 * 4;    // Wm1: K=32,  N=64
  const int T_M2 = 2 * 9;    // Wm2: K=64,  N=144
  const int T_LN = 5 * 8;    // Wln: K=144 (pad 160), N=128
  const int T_F1 = 4 * 32;   // Wf1: K=128, N=512
  const int T_F2 = 16 * 8;   // Wf2: K=512, N=128

  char* ws = (char*)d_ws;
  size_t off = 0;
  float* summed = (float*)(ws + off); off += (size_t)N_NODES * D_NF * 4;   // 57.6 MB (reused as ybuf)
  float* cnt    = (float*)(ws + off); off += (size_t)N_NODES * 4;
  float* deg    = (float*)(ws + off); off += (size_t)N_NODES * 4;
  float* stats  = (float*)(ws + off); off += 1024 * 4;
  size_t zero_bytes = off;                                                  // contiguous zero region
  float*  hbuf  = (float*)(ws + off); off += (size_t)N_NODES * D_OUT * 4;  // 51.2 MB
  half_t* Wm1f  = (half_t*)(ws + off); off += (size_t)T_M1 * 1024;
  half_t* Wm2f  = (half_t*)(ws + off); off += (size_t)T_M2 * 1024;
  half_t* Wlnf  = (half_t*)(ws + off); off += (size_t)T_LN * 1024;
  half_t* Wf1f  = (half_t*)(ws + off); off += (size_t)T_F1 * 1024;
  half_t* Wf2f  = (half_t*)(ws + off); off += (size_t)T_F2 * 1024;

  float* sum1 = stats,       *sq1 = stats + 128, *mu1 = stats + 256, *rs1 = stats + 384;
  float* sum2 = stats + 512, *sq2 = stats + 640, *mu2 = stats + 768, *rs2 = stats + 896;
  float* ybuf = summed;

  hipMemsetAsync(d_ws, 0, zero_bytes, stream);

  k_pack_frags<<<(T_M1 * 32 + 255) / 256, 256, 0, stream>>>(Wm1, Wm1f, D_EF,  D_MOD, 4,  T_M1);
  k_pack_frags<<<(T_M2 * 32 + 255) / 256, 256, 0, stream>>>(Wm2, Wm2f, D_MOD, D_NF,  9,  T_M2);
  k_pack_frags<<<(T_LN * 32 + 255) / 256, 256, 0, stream>>>(Wln, Wlnf, D_NF,  D_OUT, 8,  T_LN);
  k_pack_frags<<<(T_F1 * 32 + 255) / 256, 256, 0, stream>>>(Wf1, Wf1f, D_OUT, D_FFN, 32, T_F1);
  k_pack_frags<<<(T_F2 * 32 + 255) / 256, 256, 0, stream>>>(Wf2, Wf2f, D_FFN, D_OUT, 8,  T_F2);

  k_edge<<<(N_EDGES + 63) / 64, 128, 0, stream>>>(x, x_pe, ea, ep, eidx,
                                                  Wm1f, bm1, Wm2f, bm2, summed, cnt, deg);
  k_node_lin<<<(N_NODES + 63) / 64, 128, 0, stream>>>(summed, cnt, deg, Wlnf, bln,
                                                      th1, th2, hbuf, sum1, sq1);
  k_stats<<<1, 128, 0, stream>>>(sum1, sq1, mu1, rs1, (float)N_NODES);
  k_bn1<<<(N_NODES * D_OUT + 255) / 256, 256, 0, stream>>>(hbuf, x, mu1, rs1, g1, be1);
  k_ffn<<<(N_NODES + 63) / 64, 128, 0, stream>>>(hbuf, Wf1f, bf1, Wf2f, bf2, ybuf, sum2, sq2);
  k_stats<<<1, 128, 0, stream>>>(sum2, sq2, mu2, rs2, (float)N_NODES);
  k_bn2<<<(N_NODES * D_OUT + 255) / 256, 256, 0, stream>>>(ybuf, (float*)d_out, mu2, rs2, g2, be2);
}
